// GATBondk_89584427860009
// MI455X (gfx1250) — matrix operand order repaired and checked by reading, not compiled, not run on hardware
//
#include <hip/hip_runtime.h>
#include <hip/hip_bf16.h>
#include <cstdint>
#include <cstddef>

typedef __attribute__((ext_vector_type(16))) _Float16 v16h;
typedef __attribute__((ext_vector_type(8)))  _Float16 v8h;
typedef __attribute__((ext_vector_type(8)))  float    v8f;

#define NNODES 50000
#define NEDGES 800000
#define INF_   128
#define HID_   64

// ---------- float <-> order-preserving u32 (for atomic float max) ----------
static __device__ __forceinline__ unsigned f2ord(float f) {
  unsigned b = __float_as_uint(f);
  return (b & 0x80000000u) ? ~b : (b | 0x80000000u);
}
static __device__ __forceinline__ float ord2f(unsigned u) {
  unsigned b = (u & 0x80000000u) ? (u & 0x7fffffffu) : ~u;
  return __uint_as_float(b);
}

// ---------- WMMA f16 GEMM, row-block per wave: C[M,N] = A[M,K] @ Bt[N,K]^T ---
// A row-major f16 [M,K]; Bt row-major f16 [TN*16,K] (B transposed, N padded).
// Each wave32 owns one 16-row block of C and sweeps all TN 16-col tiles,
// so each A row block is read from HBM exactly once (A is the only large,
// un-cacheable operand: weights are <=40KB and live in WGP$/L2).
template <int TN>
__global__ __launch_bounds__(256)
void k_gemm_wmma_row(const _Float16* __restrict__ A,
                     const _Float16* __restrict__ Bt,
                     const float* __restrict__ bias,
                     float* __restrict__ Cf,
                     _Float16* __restrict__ Ch,
                     int M, int K, int N_out, int ldc, int relu)
{
  const int wave = blockIdx.x * (blockDim.x >> 5) + (threadIdx.x >> 5);
  if (wave * 16 >= M) return;               // whole-wave exit; EXEC stays full

  const int lane = threadIdx.x & 31;
  const int r    = lane & 15;               // A row / C col / Bt row (within tile)
  const int hi   = lane >> 4;               // half-wave selector

  const _Float16* __restrict__ arow = A + (size_t)(wave * 16 + r) * K;

  v8f acc[TN] = {};
  for (int k0 = 0; k0 < K; k0 += 32) {
    // A 16x32 f16 layout: lane(r,hi) holds K = k0+hi*16 .. +15 (contig, matches B)
    v16h a = *(const v16h*)(arow + k0 + hi * 16);
#pragma unroll
    for (int t = 0; t < TN; ++t) {
      // B 32x16 f16 layout: lane(n=r,hi) holds K = k0+hi*16 .. +15 (contig in Bt)
      v16h b = *(const v16h*)(Bt + (size_t)(t * 16 + r) * K + k0 + hi * 16);
      acc[t] = __builtin_amdgcn_wmma_f32_16x16x32_f16(false, a, false, b,
                                                      (short)0, acc[t], false, false);
    }
  }

#pragma unroll
  for (int t = 0; t < TN; ++t) {
    const int col = t * 16 + r;
    const float bv = (bias != nullptr && col < N_out) ? bias[col] : 0.f;
#pragma unroll
    for (int v = 0; v < 8; ++v) {
      // C/D f32 layout: VGPR v, lanes 0-15 -> M = v, lanes 16-31 -> M = v + 8
      const int row = wave * 16 + v + hi * 8;
      float val = acc[t][v] + bv;
      if (relu) val = fmaxf(val, 0.f);
      if (col < N_out) {
        const size_t off = (size_t)row * ldc + col;
        if (Cf) Cf[off] = val;
        if (Ch) Ch[off] = (_Float16)val;
      }
    }
  }
}

// ---------- small prep / elementwise kernels ----------
__global__ void k_f32_to_f16(const float* __restrict__ in, _Float16* __restrict__ out, int n) {
  int i = blockIdx.x * blockDim.x + threadIdx.x;
  if (i < n) out[i] = (_Float16)in[i];
}

// Bt[n*K+k] = W[k*N+n], zero-padded to Npad rows
__global__ void k_conv_wt(const float* __restrict__ W, _Float16* __restrict__ Bt,
                          int K, int N, int Npad) {
  int i = blockIdx.x * blockDim.x + threadIdx.x;
  if (i >= Npad * K) return;
  int n = i / K, k = i - n * K;
  Bt[i] = (n < N) ? (_Float16)W[(size_t)k * N + n] : (_Float16)0.f;
}

// v[j] = sum_c We[j][c] * ae[c]   (edge-attr attention vector, j=0,1)
__global__ void k_edge_vec(const float* __restrict__ We, const float* __restrict__ ae,
                           float* __restrict__ v) {
  int j = threadIdx.x;
  if (j < 2) {
    float s = 0.f;
    for (int c = 0; c < HID_; ++c) s += We[j * HID_ + c] * ae[c];
    v[j] = s;
  }
}

__global__ void k_fill_u32(unsigned* __restrict__ p, unsigned val, int n) {
  int i = blockIdx.x * blockDim.x + threadIdx.x;
  if (i < n) p[i] = val;
}

__global__ void k_node_scores(const float* __restrict__ h,
                              const float* __restrict__ a_s, const float* __restrict__ a_d,
                              float* __restrict__ s, float* __restrict__ d) {
  int i = blockIdx.x * blockDim.x + threadIdx.x;
  if (i >= NNODES) return;
  float ss = 0.f, dd = 0.f;
  const float* hr = h + (size_t)i * HID_;
  for (int c = 0; c < HID_; ++c) { ss += hr[c] * a_s[c]; dd += hr[c] * a_d[c]; }
  s[i] = ss; d[i] = dd;
}

__global__ void k_edge_logit(const float* __restrict__ s, const float* __restrict__ d,
                             const int* __restrict__ src, const int* __restrict__ dst,
                             const float* __restrict__ ea, const float* __restrict__ v2,
                             float* __restrict__ logit, unsigned* __restrict__ mbits) {
  int e = blockIdx.x * blockDim.x + threadIdx.x;
  if (e >= NEDGES) return;
  float lg = s[src[e]] + d[dst[e]] + ea[2 * e] * v2[0] + ea[2 * e + 1] * v2[1];
  lg = (lg > 0.f) ? lg : 0.2f * lg;                 // leaky_relu, slope 0.2
  logit[e] = lg;
  atomicMax(&mbits[dst[e]], f2ord(lg));             // segment max over dst
}

__global__ void k_edge_exp(const float* __restrict__ logit, const int* __restrict__ dst,
                           const unsigned* __restrict__ mbits,
                           float* __restrict__ ex, float* __restrict__ den) {
  int e = blockIdx.x * blockDim.x + threadIdx.x;
  if (e >= NEDGES) return;
  float m = ord2f(mbits[dst[e]]);
  float v = __expf(logit[e] - m);
  ex[e] = v;
  atomicAdd(&den[dst[e]], v);                       // segment sum
}

// one wave per edge; each lane scatters 2 of the 64 channels
__global__ void k_edge_agg(const float* __restrict__ h,
                           const int* __restrict__ src, const int* __restrict__ dst,
                           const float* __restrict__ ex, const float* __restrict__ den,
                           float* __restrict__ agg) {
  int t = blockIdx.x * blockDim.x + threadIdx.x;
  int e = t >> 5;
  if (e >= NEDGES) return;
  int c = (t & 31) * 2;
  float al = ex[e] / fmaxf(den[dst[e]], 1e-16f);
  const size_t sb = (size_t)src[e] * HID_, db = (size_t)dst[e] * HID_;
  atomicAdd(&agg[db + c],     al * h[sb + c]);
  atomicAdd(&agg[db + c + 1], al * h[sb + c + 1]);
}

__global__ void k_node_finish(const float* __restrict__ agg, const float* __restrict__ bias,
                              float* __restrict__ yf, _Float16* __restrict__ yh) {
  int i = blockIdx.x * blockDim.x + threadIdx.x;
  if (i >= NNODES * HID_) return;
  int c = i & (HID_ - 1);
  float v = fmaxf(agg[i] + bias[c], 0.f);           // relu(gat + bias)
  yf[i] = v;
  if (yh) yh[i] = (_Float16)v;
}

__global__ void k_add_to_f16(const float* __restrict__ a, const float* __restrict__ b,
                             _Float16* __restrict__ o, int n) {
  int i = blockIdx.x * blockDim.x + threadIdx.x;
  if (i < n) o[i] = (_Float16)(a[i] + b[i]);
}

// xb[e] = concat( y4[src]+y4[dst] , relu(edge_attr @ We0 + be0) )  in f16
__global__ void k_build_xb(const float* __restrict__ y4,
                           const int* __restrict__ src, const int* __restrict__ dst,
                           const float* __restrict__ ea,
                           const float* __restrict__ We0, const float* __restrict__ be0,
                           _Float16* __restrict__ xb) {
  long long t = (long long)blockIdx.x * blockDim.x + threadIdx.x;
  if (t >= (long long)NEDGES * HID_) return;
  int e = (int)(t >> 6);
  int c = (int)(t & 63);
  xb[(size_t)e * 128 + c] =
      (_Float16)(y4[(size_t)src[e] * HID_ + c] + y4[(size_t)dst[e] * HID_ + c]);
  float v = ea[2 * e] * We0[c] + ea[2 * e + 1] * We0[HID_ + c] + be0[c];
  xb[(size_t)e * 128 + 64 + c] = (_Float16)fmaxf(v, 0.f);
}

// ======================= host-side orchestration =======================
extern "C" void kernel_launch(void* const* d_in, const int* in_sizes, int n_in,
                              void* d_out, int out_size, void* d_ws, size_t ws_size,
                              hipStream_t stream) {
  (void)in_sizes; (void)n_in; (void)out_size; (void)ws_size;

  const float* x   = (const float*)d_in[0];
  const int*   ei  = (const int*)d_in[1];
  const float* ea  = (const float*)d_in[2];
  const float* Wg1 = (const float*)d_in[4];
  const float* as1 = (const float*)d_in[5];
  const float* ad1 = (const float*)d_in[6];
  const float* We1 = (const float*)d_in[7];
  const float* ae1 = (const float*)d_in[8];
  const float* bg1 = (const float*)d_in[9];
  const float* Wg2 = (const float*)d_in[10];
  const float* as2 = (const float*)d_in[11];
  const float* ad2 = (const float*)d_in[12];
  const float* We2 = (const float*)d_in[13];
  const float* ae2 = (const float*)d_in[14];
  const float* bg2 = (const float*)d_in[15];
  const float* W2  = (const float*)d_in[16];
  const float* b2  = (const float*)d_in[17];
  const float* We0 = (const float*)d_in[18];
  const float* be0 = (const float*)d_in[19];
  const float* W3  = (const float*)d_in[20];
  const float* b3  = (const float*)d_in[21];
  const float* Wm0 = (const float*)d_in[22];
  const float* bm0 = (const float*)d_in[23];
  const float* Wm1 = (const float*)d_in[24];
  const float* bm1 = (const float*)d_in[25];
  const float* Wm2 = (const float*)d_in[26];
  const float* bm2 = (const float*)d_in[27];
  const float* W4  = (const float*)d_in[28];
  const float* b4  = (const float*)d_in[29];

  const int* src = ei;
  const int* dst = ei + NEDGES;

  // ---- carve workspace ----
  size_t off = 0;
  auto alloc = [&](size_t bytes) -> void* {
    void* p = (char*)d_ws + off;
    off += (bytes + 255) & ~(size_t)255;
    return p;
  };
  _Float16* wg1t = (_Float16*)alloc((size_t)HID_ * INF_ * 2);
  _Float16* wg2t = (_Float16*)alloc((size_t)HID_ * HID_ * 2);
  _Float16* w2t  = (_Float16*)alloc((size_t)HID_ * HID_ * 2);
  _Float16* w3t  = (_Float16*)alloc((size_t)128 * 128 * 2);
  _Float16* wm0t = (_Float16*)alloc((size_t)128 * 128 * 2);
  _Float16* wm1t = (_Float16*)alloc((size_t)128 * 128 * 2);
  _Float16* wm2t = (_Float16*)alloc((size_t)128 * 128 * 2);
  _Float16* w4t  = (_Float16*)alloc((size_t)160 * 128 * 2);   // 145 padded -> 160
  float*    v1   = (float*)alloc(2 * sizeof(float));
  float*    v2g  = (float*)alloc(2 * sizeof(float));
  _Float16* xh   = (_Float16*)alloc((size_t)NNODES * INF_ * 2);
  float*    hbuf = (float*)alloc((size_t)NNODES * HID_ * 4);
  float*    sbuf = (float*)alloc((size_t)NNODES * 4);
  float*    dbuf = (float*)alloc((size_t)NNODES * 4);
  unsigned* mbit = (unsigned*)alloc((size_t)NNODES * 4);
  float*    den  = (float*)alloc((size_t)NNODES * 4);
  float*    lgt  = (float*)alloc((size_t)NEDGES * 4);
  float*    exb  = (float*)alloc((size_t)NEDGES * 4);
  float*    agg  = (float*)alloc((size_t)NNODES * HID_ * 4);
  float*    y0f  = (float*)alloc((size_t)NNODES * HID_ * 4);
  _Float16* y0h  = (_Float16*)alloc((size_t)NNODES * HID_ * 2);
  float*    y1f  = (float*)alloc((size_t)NNODES * HID_ * 4);
  _Float16* s01h = (_Float16*)alloc((size_t)NNODES * HID_ * 2);
  float*    y4f  = (float*)alloc((size_t)NNODES * HID_ * 4);
  _Float16* xb   = (_Float16*)alloc((size_t)NEDGES * 128 * 2);
  _Float16* yb0  = (_Float16*)alloc((size_t)NEDGES * 128 * 2);
  _Float16* yb1  = (_Float16*)alloc((size_t)NEDGES * 128 * 2);

  auto cdiv = [](long long a, long long b) -> int { return (int)((a + b - 1) / b); };

  // ---- weight prep (f16 transposed, padded) ----
  k_conv_wt<<<cdiv((size_t)HID_ * INF_, 256), 256, 0, stream>>>(Wg1, wg1t, INF_, HID_, HID_);
  k_conv_wt<<<cdiv((size_t)HID_ * HID_, 256), 256, 0, stream>>>(Wg2, wg2t, HID_, HID_, HID_);
  k_conv_wt<<<cdiv((size_t)HID_ * HID_, 256), 256, 0, stream>>>(W2,  w2t,  HID_, HID_, HID_);
  k_conv_wt<<<cdiv(128 * 128, 256), 256, 0, stream>>>(W3,  w3t,  128, 128, 128);
  k_conv_wt<<<cdiv(128 * 128, 256), 256, 0, stream>>>(Wm0, wm0t, 128, 128, 128);
  k_conv_wt<<<cdiv(128 * 128, 256), 256, 0, stream>>>(Wm1, wm1t, 128, 128, 128);
  k_conv_wt<<<cdiv(128 * 128, 256), 256, 0, stream>>>(Wm2, wm2t, 128, 128, 128);
  k_conv_wt<<<cdiv(160 * 128, 256), 256, 0, stream>>>(W4,  w4t,  128, 145, 160);
  k_edge_vec<<<1, 32, 0, stream>>>(We1, ae1, v1);
  k_edge_vec<<<1, 32, 0, stream>>>(We2, ae2, v2g);
  k_f32_to_f16<<<cdiv((size_t)NNODES * INF_, 256), 256, 0, stream>>>(x, xh, NNODES * INF_);

  // waves = M/16 row blocks; 8 waves per 256-thread block
  auto gemm = [&](const _Float16* A, const _Float16* Bt, const float* bias,
                  float* Cf, _Float16* Ch, int M, int Npad, int K, int Nout, int ldc, int relu) {
    int blocks = cdiv(M / 16, 8);
    switch (Npad) {
      case 64:
        k_gemm_wmma_row<4><<<blocks, 256, 0, stream>>>(A, Bt, bias, Cf, Ch, M, K, Nout, ldc, relu);
        break;
      case 128:
        k_gemm_wmma_row<8><<<blocks, 256, 0, stream>>>(A, Bt, bias, Cf, Ch, M, K, Nout, ldc, relu);
        break;
      case 160:
        k_gemm_wmma_row<10><<<blocks, 256, 0, stream>>>(A, Bt, bias, Cf, Ch, M, K, Nout, ldc, relu);
        break;
    }
  };

  auto gat_layer = [&](const _Float16* Ah, int Kin, const _Float16* Wt,
                       const float* a_s, const float* a_d, const float* vv,
                       const float* bias, float* outf, _Float16* outh) {
    gemm(Ah, Wt, nullptr, hbuf, nullptr, NNODES, HID_, Kin, HID_, HID_, 0);      // h = x @ W
    k_node_scores<<<cdiv(NNODES, 256), 256, 0, stream>>>(hbuf, a_s, a_d, sbuf, dbuf);
    k_fill_u32<<<cdiv(NNODES, 256), 256, 0, stream>>>(mbit, 0u, NNODES);
    k_fill_u32<<<cdiv(NNODES, 256), 256, 0, stream>>>((unsigned*)den, 0u, NNODES);
    k_fill_u32<<<cdiv((size_t)NNODES * HID_, 256), 256, 0, stream>>>((unsigned*)agg, 0u, NNODES * HID_);
    k_edge_logit<<<cdiv(NEDGES, 256), 256, 0, stream>>>(sbuf, dbuf, src, dst, ea, vv, lgt, mbit);
    k_edge_exp<<<cdiv(NEDGES, 256), 256, 0, stream>>>(lgt, dst, mbit, exb, den);
    k_edge_agg<<<cdiv((long long)NEDGES * 32, 256), 256, 0, stream>>>(hbuf, src, dst, exb, den, agg);
    k_node_finish<<<cdiv((size_t)NNODES * HID_, 256), 256, 0, stream>>>(agg, bias, outf, outh);
  };

  // ---- GAT layers ----
  gat_layer(xh,  INF_, wg1t, as1, ad1, v1,  bg1, y0f, y0h);     // y0
  gat_layer(y0h, HID_, wg2t, as2, ad2, v2g, bg2, y1f, nullptr); // y1

  // ---- y4 = relu((y0+y1) @ W2 + b2) ----
  k_add_to_f16<<<cdiv((size_t)NNODES * HID_, 256), 256, 0, stream>>>(y0f, y1f, s01h, NNODES * HID_);
  gemm(s01h, w2t, b2, y4f, nullptr, NNODES, HID_, HID_, HID_, HID_, 1);

  // ---- per-edge features xb = [y4[src]+y4[dst], relu(ea@We0+be0)] ----
  k_build_xb<<<cdiv((long long)NEDGES * HID_, 256), 256, 0, stream>>>(y4f, src, dst, ea, We0, be0, xb);

  // ---- edge MLP (the 134-GFLOP part), all WMMA ----
  gemm(xb,  w3t,  b3,  nullptr, yb0, NEDGES, 128, 128, 128, 128, 1);
  gemm(yb0, wm0t, bm0, nullptr, yb1, NEDGES, 128, 128, 128, 128, 1);
  gemm(yb1, wm1t, bm1, nullptr, xb,  NEDGES, 128, 128, 128, 128, 1);
  gemm(xb,  wm2t, bm2, nullptr, yb0, NEDGES, 128, 128, 128, 128, 1);
  gemm(yb0, w4t,  b4,  (float*)d_out, nullptr, NEDGES, 160, 128, 145, 145, 0);
}
